// GConvMultiScale_19842748908317
// MI455X (gfx1250) — matrix operand order repaired and checked by reading, not compiled, not run on hardware
//
#include <hip/hip_runtime.h>

#define NN 50000
#define NE 800000
#define NS 4
#define D  128

typedef __attribute__((ext_vector_type(8)))  float    v8f;
typedef __attribute__((ext_vector_type(2)))  float    v2f;
typedef __attribute__((ext_vector_type(16))) _Float16 v16h;

#if defined(__AMDGCN__)
#if __has_builtin(__builtin_amdgcn_wmma_f32_16x16x4_f32)
#define USE_F32_WMMA 1
#endif
#endif

__device__ __forceinline__ void atomAddF(float* p, float v) {
#if defined(__AMDGCN__)
  unsafeAtomicAdd(p, v);   // native global_atomic_add_f32 (no return)
#else
  atomicAdd(p, v);
#endif
}

// ---------------- per-scale CSR build ----------------
__global__ void count_deg_k(const int* __restrict__ dst, const float* __restrict__ ew,
                            int* __restrict__ cnt, float* __restrict__ deg, int E) {
  int e = blockIdx.x * blockDim.x + threadIdx.x;
  if (e < E) {
    const int d = dst[e];
    atomicAdd(&cnt[d], 1);
    atomAddF(&deg[d], ew[e]);
  }
}

__global__ void dinv_k(float* __restrict__ deg, int n) {
  int i = blockIdx.x * blockDim.x + threadIdx.x;
  if (i < n) deg[i] = rsqrtf(deg[i] + 1.0f);  // +1 self loop; always > 0
}

// single-block exclusive scan over 50k counts (LDS Hillis-Steele, carried chunks)
__global__ __launch_bounds__(1024) void scan_k(const int* __restrict__ cnt,
                                               int* __restrict__ off,
                                               int* __restrict__ cur, int n) {
  __shared__ int tmp[1024];
  __shared__ int carry;
  if (threadIdx.x == 0) carry = 0;
  __syncthreads();
  for (int base = 0; base < n; base += 1024) {
    const int i = base + (int)threadIdx.x;
    const int v = (i < n) ? cnt[i] : 0;
    tmp[threadIdx.x] = v;
    __syncthreads();
#pragma unroll
    for (int s = 1; s < 1024; s <<= 1) {
      const int t = (threadIdx.x >= (unsigned)s) ? tmp[threadIdx.x - s] : 0;
      __syncthreads();
      tmp[threadIdx.x] += t;
      __syncthreads();
    }
    const int excl = tmp[threadIdx.x] - v + carry;   // carry read before update
    if (i < n) { off[i] = excl; cur[i] = excl; }
    __syncthreads();
    if (threadIdx.x == 1023) carry += tmp[1023];
    __syncthreads();
  }
  if (threadIdx.x == 0) off[n] = carry;
}

__global__ void csr_build_k(const int* __restrict__ src, const int* __restrict__ dst,
                            const float* __restrict__ ew, int* __restrict__ cur,
                            int* __restrict__ csrc, float* __restrict__ cw, int E) {
  int e = blockIdx.x * blockDim.x + threadIdx.x;
  if (e < E) {
    const int p = atomicAdd(&cur[dst[e]], 1);
    csrc[p] = src[e];
    cw[p]   = ew[e];
  }
}

// ---------------- WMMA GEMM: H = A(N x 128) * B(128 x 128) ----------------
// one wave32 per 16x16 output tile; 8 waves / block
__global__ __launch_bounds__(256) void gemm_wmma_k(const float* __restrict__ A,
                                                   const float* __restrict__ Bm,
                                                   float* __restrict__ H,
                                                   int nRowTiles) {
  const int tile = blockIdx.x * 8 + threadIdx.y;
  const int rowTile = tile >> 3;       // D/16 = 8 col tiles
  const int colTile = tile & 7;
  if (rowTile >= nRowTiles) return;
  const int lane = threadIdx.x;        // 0..31
  const int half = lane >> 4;
  const int r    = lane & 15;
  const int row  = rowTile * 16 + r;
  const int col  = colTile * 16 + r;
  const float* arow = A + (size_t)row * D;
  v8f c = {};
#if defined(USE_F32_WMMA)
  for (int k = 0; k < D; k += 4) {
    const int ka = k + half * 2;       // A 16x4 f32 layout: lanes 0-15 K=0,1; 16-31 K=2,3
    v2f a; a.x = arow[ka];                 a.y = arow[ka + 1];
    v2f b; b.x = Bm[(size_t)ka * D + col]; b.y = Bm[(size_t)(ka + 1) * D + col];
    c = __builtin_amdgcn_wmma_f32_16x16x4_f32(false, a, false, b, (short)0, c,
                                              false, false);
  }
#else
  for (int k = 0; k < D; k += 32) {
    v16h a, b;
    const int ab = k + half * 8;
#pragma unroll
    for (int j = 0; j < 8; ++j) {
      a[j]     = (_Float16)arow[ab + j];
      a[8 + j] = (_Float16)arow[ab + 16 + j];
    }
    const int bb = k + half * 8;
#pragma unroll
    for (int j = 0; j < 16; ++j) b[j] = (_Float16)Bm[(size_t)(bb + j + (j >= 8 ? 8 : 0)) * D + col];
    c = __builtin_amdgcn_wmma_f32_16x16x32_f16(false, a, false, b, (short)0, c,
                                               false, false);
  }
#endif
  // C/D 16x16 f32 layout: VGPR v -> M = v + half*8, N = r
  float* hrow = H + ((size_t)rowTile * 16 + half * 8) * D + colTile * 16 + r;
#pragma unroll
  for (int v = 0; v < 8; ++v) hrow[(size_t)v * D] = c[v];
}

// ------- fused gather + self loop + bias + PReLU: one wave32 per node -------
__global__ __launch_bounds__(256) void aggregate_k(const int* __restrict__ off,
                                                   const int* __restrict__ csrc,
                                                   const float* __restrict__ cw,
                                                   const float* __restrict__ dinv,
                                                   const float* __restrict__ h,
                                                   const float* __restrict__ bias,
                                                   const float* __restrict__ pa,
                                                   float* __restrict__ out, int n) {
  const int node = blockIdx.x * 8 + (int)(threadIdx.x >> 5);
  if (node >= n) return;
  const int lane = threadIdx.x & 31;
  const float dd = dinv[node];
  float4 acc;
  {
    const float di2 = dd * dd;  // self loop, weight 1: dinv*1*dinv
    const float4 hh = ((const float4*)(h + (size_t)node * D))[lane];
    acc.x = di2 * hh.x; acc.y = di2 * hh.y; acc.z = di2 * hh.z; acc.w = di2 * hh.w;
  }
  const int e0 = off[node], e1 = off[node + 1];
  for (int e = e0; e < e1; ++e) {
    const int   s = csrc[e];
    const float w = cw[e] * dinv[s] * dd;
    const float4 hs = ((const float4*)(h + (size_t)s * D))[lane];
    acc.x += w * hs.x; acc.y += w * hs.y; acc.z += w * hs.z; acc.w += w * hs.w;
  }
  const float4 bb = ((const float4*)bias)[lane];
  const float4 aa = ((const float4*)pa)[lane];
  float4 r;
  r.x = acc.x + bb.x;  r.x = r.x > 0.f ? r.x : aa.x * r.x;
  r.y = acc.y + bb.y;  r.y = r.y > 0.f ? r.y : aa.y * r.y;
  r.z = acc.z + bb.z;  r.z = r.z > 0.f ? r.z : aa.z * r.z;
  r.w = acc.w + bb.w;  r.w = r.w > 0.f ? r.w : aa.w * r.w;
  ((float4*)(out + (size_t)node * D))[lane] = r;
}

extern "C" void kernel_launch(void* const* d_in, const int* in_sizes, int n_in,
                              void* d_out, int out_size, void* d_ws, size_t ws_size,
                              hipStream_t stream) {
  (void)in_sizes; (void)n_in; (void)out_size; (void)ws_size;
  const float* x  = (const float*)d_in[0];
  const int*   ei = (const int*)d_in[1];     // [NS][2][NE]
  const float* ew = (const float*)d_in[2];   // [NS][NE]
  const float* W0 = (const float*)d_in[3];
  const float* b0 = (const float*)d_in[4];
  const float* W1 = (const float*)d_in[5];
  const float* b1 = (const float*)d_in[6];
  const float* pa = (const float*)d_in[7];
  float* out = (float*)d_out;

  // workspace layout (16B-aligned h/z)
  float* dinv = (float*)d_ws;                 // NN
  int*   cnt  = (int*)(dinv + NN);            // NN
  int*   off  = cnt + NN;                     // NN + 16 (padded for alignment)
  int*   cur  = off + NN + 16;                // NN
  int*   csrc = cur + NN;                     // NE
  float* cw   = (float*)(csrc + NE);          // NE
  float* h    = cw + NE;                      // NN*D
  float* z    = h + (size_t)NN * D;           // NN*D

  const int nRowTiles = NN / 16;              // 3125 exactly
  const dim3 gemmBlock(32, 8);
  const int  gemmGrid = nRowTiles;            // 8 tiles (one row strip) per block
  const int  edgeGrid = (NE + 255) / 256;
  const int  aggGrid  = (NN + 7) / 8;

  for (int t = 0; t < NS; ++t) {
    const int*   srcI = ei + (size_t)t * 2 * NE;
    const int*   dstI = srcI + NE;
    const float* ewt  = ew + (size_t)t * NE;

    // CSR + normalization for this scale (shared by both layers)
    hipMemsetAsync(dinv, 0, (size_t)NN * sizeof(float), stream);
    hipMemsetAsync(cnt,  0, (size_t)NN * sizeof(int), stream);
    count_deg_k<<<edgeGrid, 256, 0, stream>>>(dstI, ewt, cnt, dinv, NE);
    dinv_k<<<(NN + 255) / 256, 256, 0, stream>>>(dinv, NN);
    scan_k<<<1, 1024, 0, stream>>>(cnt, off, cur, NN);
    csr_build_k<<<edgeGrid, 256, 0, stream>>>(srcI, dstI, ewt, cur, csrc, cw, NE);

    // ---- layer 1: z = prelu(gcn(x, W0, b0)) ----
    gemm_wmma_k<<<gemmGrid, gemmBlock, 0, stream>>>(x, W0, h, nRowTiles);
    aggregate_k<<<aggGrid, 256, 0, stream>>>(off, csrc, cw, dinv, h, b0, pa, z, NN);

    // ---- layer 2: out[t] = prelu(gcn(z, W1, b1)) ----
    gemm_wmma_k<<<gemmGrid, gemmBlock, 0, stream>>>(z, W1, h, nRowTiles);
    aggregate_k<<<aggGrid, 256, 0, stream>>>(off, csrc, cw, dinv, h, b1, pa,
                                             out + (size_t)t * NN * D, NN);
  }
}
